// EGNNModel_cpl_global_7885559956065
// MI455X (gfx1250) — matrix operand order repaired and checked by reading, not compiled, not run on hardware
//
#include <hip/hip_runtime.h>
#include <hip/hip_bf16.h>

#define NN 50000
#define EE 800000

typedef __attribute__((ext_vector_type(16))) _Float16 v16h;
typedef __attribute__((ext_vector_type(8)))  _Float16 v8h;
typedef __attribute__((ext_vector_type(8)))  float    v8f;

__device__ __forceinline__ float silu_f(float x) { return x * (1.0f / (1.0f + __expf(-x))); }

// Wave-level LDS ordering fence (CDNA5 split dependency counters)
__device__ __forceinline__ void lds_fence() { asm volatile("s_wait_dscnt 0" ::: "memory"); }
// Async (global->LDS) completion fence
__device__ __forceinline__ void async_fence() { asm volatile("s_wait_asynccnt 0" ::: "memory"); }

// Per-lane async copy of 16B global -> 16B LDS (ASYNCcnt tracked).
// Low 32 bits of a generic shared pointer are the LDS offset (aperture rules).
__device__ __forceinline__ void async_to_lds_b128(void* lds, const void* g) {
  unsigned off = (unsigned)(unsigned long long)(uintptr_t)lds;
  asm volatile("global_load_async_to_lds_b128 %0, %1, off"
               :: "v"(off), "v"(g) : "memory");
}

// A fragment (16x32 f16 tile, row-major LDS, leading dim ldk halves):
// lane holds row (lane&15); 8 halves at k = kt + (lane>>4)*8 and 8 at k+16.
__device__ __forceinline__ v16h loadA16(const _Float16* base, int ldk, int lane, int kt) {
  const int m = lane & 15, koff = (lane >> 4) * 8;
  const _Float16* p = base + m * ldk + kt + koff;
  v8h lo = *(const v8h*)p;
  v8h hi = *(const v8h*)(p + 16);
  v16h r;
#pragma unroll
  for (int i = 0; i < 8; ++i) { r[i] = lo[i]; r[i + 8] = hi[i]; }
  return r;
}

// B fragment (32x16 f16 tile) from transposed weights Wt[n][k] in LDS:
// lane holds column (lane&15); 8 halves at k = kt + (lane>>4)*8 and 8 at k+16.
__device__ __forceinline__ v16h loadB16(const _Float16* wt, int ldk, int lane, int nt, int kt) {
  const int n = lane & 15, koff = (lane >> 4) * 8;
  const _Float16* p = wt + (nt * 16 + n) * ldk + kt + koff;
  v8h lo = *(const v8h*)p;
  v8h hi = *(const v8h*)(p + 16);
  v16h r;
#pragma unroll
  for (int i = 0; i < 8; ++i) { r[i] = lo[i]; r[i + 8] = hi[i]; }
  return r;
}

#define WMMA_F16(a, b, c) \
  __builtin_amdgcn_wmma_f32_16x16x32_f16(false, (a), false, (b), (short)0, (c), false, false)

// ---------------------------------------------------------------------------
// Edge kernel: one wave = 16 edges. Builds h=[feat[r]|feat[c]|ea|d2] (K padded
// 131->160) with async global->LDS f16 gathers, runs msg MLP + pos MLP via
// WMMA, scatters msg and diff*w with f32 atomics (L2-resident accumulators).
// ---------------------------------------------------------------------------
__global__ void __launch_bounds__(256) egnn_edge_kernel(
    const _Float16* __restrict__ feat, const float* __restrict__ pos,
    const float* __restrict__ eattr,
    const int* __restrict__ rowI, const int* __restrict__ colI,
    const float* __restrict__ mW1, const float* __restrict__ mb1,
    const float* __restrict__ mW2, const float* __restrict__ mb2,
    const float* __restrict__ pW1, const float* __restrict__ pb1,
    const float* __restrict__ pW2, const float* __restrict__ pb2,
    float* __restrict__ msgAgg, float* __restrict__ posAgg)
{
  __shared__ alignas(16) _Float16 sW1[64 * 160];  // msg_W1^T, K padded to 160
  __shared__ alignas(16) _Float16 sW2[64 * 64];   // msg_W2^T
  __shared__ alignas(16) _Float16 sP1[64 * 64];   // pos_W1^T
  __shared__ float sPW2[64];
  __shared__ float sB1[64], sB2[64], sPB1[64];
  __shared__ alignas(16) _Float16 sA[8][16 * 160];  // per-wave A tile
  __shared__ alignas(16) _Float16 sM[8][16 * 64];   // per-wave activation tile

  const int tid = threadIdx.x;
  for (int idx = tid; idx < 64 * 160; idx += 256) {
    const int n = idx / 160, k = idx % 160;
    sW1[idx] = (_Float16)((k < 131) ? mW1[k * 64 + n] : 0.0f);
  }
  for (int idx = tid; idx < 64 * 64; idx += 256) {
    const int n = idx >> 6, k = idx & 63;
    sW2[idx] = (_Float16)mW2[k * 64 + n];
    sP1[idx] = (_Float16)pW1[k * 64 + n];
  }
  if (tid < 64) {
    sPW2[tid] = pW2[tid];
    sB1[tid] = mb1[tid];
    sB2[tid] = mb2[tid];
    sPB1[tid] = pb1[tid];
  }
  __syncthreads();

  const int wave = tid >> 5, lane = tid & 31;
  const int m16 = lane & 15, half = lane >> 4;
  const int eBase = blockIdx.x * 128 + wave * 16;

  // ---- gather phase: async f16 feature rows straight into the LDS A tile
  const long long e = eBase + m16;
  const int r = rowI[e], c = colI[e];

  _Float16* A = &sA[wave][m16 * 160];
  const _Float16* fr = feat + (long long)r * 64 + half * 32;
  const _Float16* fc = feat + (long long)c * 64 + half * 32;
#pragma unroll
  for (int i = 0; i < 4; ++i) {
    async_to_lds_b128(A + half * 32 + i * 8, fr + i * 8);
    async_to_lds_b128(A + 64 + half * 32 + i * 8, fc + i * 8);
  }

  const float dx = pos[(long long)r * 3 + 0] - pos[(long long)c * 3 + 0];
  const float dy = pos[(long long)r * 3 + 1] - pos[(long long)c * 3 + 1];
  const float dz = pos[(long long)r * 3 + 2] - pos[(long long)c * 3 + 2];
  const float d2 = dx * dx + dy * dy + dz * dz;
  if (half == 0) {
    A[128] = (_Float16)eattr[e * 2 + 0];
    A[129] = (_Float16)eattr[e * 2 + 1];
    A[130] = (_Float16)d2;
#pragma unroll
    for (int i = 131; i < 160; ++i) A[i] = (_Float16)0.0f;
  }
  // scatter row indices for this lane's accumulator rows (C layout)
  int ridx[8];
#pragma unroll
  for (int i = 0; i < 8; ++i) ridx[i] = rowI[eBase + half * 8 + i];
  async_fence();
  lds_fence();

  // ---- msg MLP layer 1: [16x160] @ [160x64], SiLU
  {
    v8f acc[4] = {};
    for (int kt = 0; kt < 160; kt += 32) {
      v16h a = loadA16(&sA[wave][0], 160, lane, kt);
#pragma unroll
      for (int t = 0; t < 4; ++t) {
        v16h b = loadB16(sW1, 160, lane, t, kt);
        acc[t] = WMMA_F16(a, b, acc[t]);
      }
    }
#pragma unroll
    for (int t = 0; t < 4; ++t) {
      const int n = t * 16 + m16;
      const float bias = sB1[n];
#pragma unroll
      for (int rr = 0; rr < 8; ++rr) {
        const int M = rr + half * 8;
        sM[wave][M * 64 + n] = (_Float16)silu_f(acc[t][rr] + bias);
      }
    }
  }
  lds_fence();

  // ---- msg MLP layer 2: [16x64] @ [64x64], SiLU; scatter msg
  {
    v8f acc[4] = {};
    for (int kt = 0; kt < 64; kt += 32) {
      v16h a = loadA16(&sM[wave][0], 64, lane, kt);
#pragma unroll
      for (int t = 0; t < 4; ++t) {
        v16h b = loadB16(sW2, 64, lane, t, kt);
        acc[t] = WMMA_F16(a, b, acc[t]);
      }
    }
    _Float16* A1 = &sA[wave][0];  // reuse A tile as [16][64] msg buffer
#pragma unroll
    for (int t = 0; t < 4; ++t) {
      const int n = t * 16 + m16;
      const float bias = sB2[n];
#pragma unroll
      for (int rr = 0; rr < 8; ++rr) {
        const int M = rr + half * 8;
        const float v = silu_f(acc[t][rr] + bias);
        A1[M * 64 + n] = (_Float16)v;
        atomicAdd(&msgAgg[(long long)ridx[rr] * 64 + n], v);
      }
    }
  }
  lds_fence();

  // ---- pos MLP layer 1: [16x64] @ [64x64], SiLU
  {
    v8f acc[4] = {};
    for (int kt = 0; kt < 64; kt += 32) {
      v16h a = loadA16(&sA[wave][0], 64, lane, kt);
#pragma unroll
      for (int t = 0; t < 4; ++t) {
        v16h b = loadB16(sP1, 64, lane, t, kt);
        acc[t] = WMMA_F16(a, b, acc[t]);
      }
    }
#pragma unroll
    for (int t = 0; t < 4; ++t) {
      const int n = t * 16 + m16;
      const float bias = sPB1[n];
#pragma unroll
      for (int rr = 0; rr < 8; ++rr) {
        const int M = rr + half * 8;
        sM[wave][M * 64 + n] = (_Float16)silu_f(acc[t][rr] + bias);
      }
    }
  }
  lds_fence();

  // ---- w = pos1 @ pW2 + pb2 (64->1, VALU) ; scatter diff*w
  {
    float part = 0.0f;
    const _Float16* m = &sM[wave][m16 * 64 + half * 32];
#pragma unroll
    for (int i = 0; i < 32; ++i) part += (float)m[i] * sPW2[half * 32 + i];
    part += __shfl_xor(part, 16, 32);
    const float w = part + pb2[0];
    if (half == 0) {
      atomicAdd(&posAgg[(long long)r * 3 + 0], dx * w);
      atomicAdd(&posAgg[(long long)r * 3 + 1], dy * w);
      atomicAdd(&posAgg[(long long)r * 3 + 2], dz * w);
    }
  }
}

// ---------------------------------------------------------------------------
// Node kernel: one wave = 16 nodes. nf_in = [feat | msg_agg*inv] (K=128),
// two WMMA GEMMs (SiLU between, none at the end) + position update.
// Out-of-range nodes are clamped on loads and predicated on stores so EXEC
// stays all-ones for WMMA.
// ---------------------------------------------------------------------------
__global__ void __launch_bounds__(256) egnn_node_kernel(
    const _Float16* __restrict__ featIn, const float* __restrict__ msgAgg,
    const float* __restrict__ invCnt,
    const float* __restrict__ posIn, const float* __restrict__ posAgg,
    const float* __restrict__ nW1, const float* __restrict__ nb1,
    const float* __restrict__ nW2, const float* __restrict__ nb2,
    _Float16* __restrict__ featOut, float* __restrict__ posOut, int updatePos)
{
  __shared__ alignas(16) _Float16 sW1[64 * 128];  // nf_W1^T
  __shared__ alignas(16) _Float16 sW2[64 * 64];   // nf_W2^T
  __shared__ float sB1[64], sB2[64];
  __shared__ alignas(16) _Float16 sA[8][16 * 128];
  __shared__ alignas(16) _Float16 sH[8][16 * 64];

  const int tid = threadIdx.x;
  for (int idx = tid; idx < 64 * 128; idx += 256) {
    const int n = idx >> 7, k = idx & 127;
    sW1[idx] = (_Float16)nW1[k * 64 + n];
  }
  for (int idx = tid; idx < 64 * 64; idx += 256) {
    const int n = idx >> 6, k = idx & 63;
    sW2[idx] = (_Float16)nW2[k * 64 + n];
  }
  if (tid < 64) { sB1[tid] = nb1[tid]; sB2[tid] = nb2[tid]; }
  __syncthreads();

  const int wave = tid >> 5, lane = tid & 31;
  const int m16 = lane & 15, half = lane >> 4;
  const int nBase = blockIdx.x * 128 + wave * 16;

  const int node = nBase + m16;
  const int g = (node < NN) ? node : (NN - 1);
  const float inv = invCnt[g];

  _Float16* A = &sA[wave][m16 * 128];
  const _Float16* fi = featIn + (long long)g * 64 + half * 32;
  const float* ma = msgAgg + (long long)g * 64 + half * 32;
#pragma unroll
  for (int i = 0; i < 4; ++i) async_to_lds_b128(A + half * 32 + i * 8, fi + i * 8);
#pragma unroll
  for (int i = 0; i < 32; ++i) A[64 + half * 32 + i] = (_Float16)(ma[i] * inv);
  async_fence();
  lds_fence();

  // [16x128] @ [128x64], SiLU
  {
    v8f acc[4] = {};
    for (int kt = 0; kt < 128; kt += 32) {
      v16h a = loadA16(&sA[wave][0], 128, lane, kt);
#pragma unroll
      for (int t = 0; t < 4; ++t) {
        v16h b = loadB16(sW1, 128, lane, t, kt);
        acc[t] = WMMA_F16(a, b, acc[t]);
      }
    }
#pragma unroll
    for (int t = 0; t < 4; ++t) {
      const int n = t * 16 + m16;
      const float bias = sB1[n];
#pragma unroll
      for (int rr = 0; rr < 8; ++rr) {
        const int M = rr + half * 8;
        sH[wave][M * 64 + n] = (_Float16)silu_f(acc[t][rr] + bias);
      }
    }
  }
  lds_fence();

  // [16x64] @ [64x64] + bias (no activation), write f16 features
  {
    v8f acc[4] = {};
    for (int kt = 0; kt < 64; kt += 32) {
      v16h a = loadA16(&sH[wave][0], 64, lane, kt);
#pragma unroll
      for (int t = 0; t < 4; ++t) {
        v16h b = loadB16(sW2, 64, lane, t, kt);
        acc[t] = WMMA_F16(a, b, acc[t]);
      }
    }
#pragma unroll
    for (int t = 0; t < 4; ++t) {
      const int n = t * 16 + m16;
      const float bias = sB2[n];
#pragma unroll
      for (int rr = 0; rr < 8; ++rr) {
        const int M = rr + half * 8;
        const int nd = nBase + M;
        if (nd < NN) featOut[(long long)nd * 64 + n] = (_Float16)(acc[t][rr] + bias);
      }
    }
  }

  if (half == 0 && node < NN) {
#pragma unroll
    for (int i = 0; i < 3; ++i) {
      float p = posIn[(long long)node * 3 + i];
      if (updatePos) p += posAgg[(long long)node * 3 + i] * inv;
      posOut[(long long)node * 3 + i] = p;
    }
  }
}

// ---------------------------------------------------------------------------
// Small utility kernels
// ---------------------------------------------------------------------------
__global__ void zero_kernel(float* p, long long n) {
  const long long i = (long long)blockIdx.x * blockDim.x + threadIdx.x;
  if (i < n) p[i] = 0.0f;
}

__global__ void embed_kernel(const float* __restrict__ nf, const float* __restrict__ W,
                             const float* __restrict__ b, _Float16* __restrict__ out) {
  const long long i = (long long)blockIdx.x * blockDim.x + threadIdx.x;
  if (i >= (long long)NN * 64) return;
  const int n = (int)(i >> 6), c = (int)(i & 63);
  out[i] = (_Float16)(nf[n * 2 + 0] * W[c] + nf[n * 2 + 1] * W[64 + c] + b[c]);
}

__global__ void count_kernel(const int* __restrict__ rowI, float* __restrict__ cnt) {
  const int e = blockIdx.x * blockDim.x + threadIdx.x;
  if (e < EE) atomicAdd(&cnt[rowI[e]], 1.0f);
}

__global__ void inv_kernel(float* cnt) {
  const int n = blockIdx.x * blockDim.x + threadIdx.x;
  if (n < NN) cnt[n] = 1.0f / fmaxf(cnt[n], 1.0f);
}

// ---------------------------------------------------------------------------
extern "C" void kernel_launch(void* const* d_in, const int* in_sizes, int n_in,
                              void* d_out, int out_size, void* d_ws, size_t ws_size,
                              hipStream_t stream) {
  const float* node_feat = (const float*)d_in[0];
  const float* node_pos  = (const float*)d_in[1];
  const float* edge_attr = (const float*)d_in[2];
  const int*   rowI      = (const int*)d_in[3];
  const int*   colI      = (const int*)d_in[4];
  const float* emb_W     = (const float*)d_in[5];
  const float* emb_b     = (const float*)d_in[6];
  const float* msg_W1    = (const float*)d_in[7];
  const float* msg_b1    = (const float*)d_in[8];
  const float* msg_W2    = (const float*)d_in[9];
  const float* msg_b2    = (const float*)d_in[10];
  const float* pos_W1    = (const float*)d_in[11];
  const float* pos_b1    = (const float*)d_in[12];
  const float* pos_W2    = (const float*)d_in[13];
  const float* pos_b2    = (const float*)d_in[14];
  const float* nf_W1     = (const float*)d_in[15];
  const float* nf_b1     = (const float*)d_in[16];
  const float* nf_W2     = (const float*)d_in[17];
  const float* nf_b2     = (const float*)d_in[18];

  char* wsb = (char*)d_ws;
  _Float16* featA = (_Float16*)wsb; wsb += (size_t)NN * 64 * sizeof(_Float16);
  _Float16* featB = (_Float16*)wsb; wsb += (size_t)NN * 64 * sizeof(_Float16);
  float* msgAgg = (float*)wsb; wsb += (size_t)NN * 64 * sizeof(float);
  float* posAgg = (float*)wsb; wsb += (size_t)NN * 3 * sizeof(float);  // contiguous after msgAgg
  float* invC   = (float*)wsb; wsb += (size_t)NN * sizeof(float);
  float* posA   = (float*)wsb; wsb += (size_t)NN * 3 * sizeof(float);
  float* posB   = (float*)wsb; wsb += (size_t)NN * 3 * sizeof(float);

  const dim3 blk(256);

  // scatter-mean denominators
  zero_kernel<<<(NN + 255) / 256, blk, 0, stream>>>(invC, NN);
  count_kernel<<<(EE + 255) / 256, blk, 0, stream>>>(rowI, invC);
  inv_kernel<<<(NN + 255) / 256, blk, 0, stream>>>(invC);

  // embedding: feat0 = (f16)(node_feat @ emb_W + emb_b)
  embed_kernel<<<(int)(((long long)NN * 64 + 255) / 256), blk, 0, stream>>>(
      node_feat, emb_W, emb_b, featA);

  const _Float16* fin = featA;
  const float* pin = node_pos;
  for (int l = 0; l < 5; ++l) {
    zero_kernel<<<(int)(((long long)NN * 67 + 255) / 256), blk, 0, stream>>>(
        msgAgg, (long long)NN * 67);

    egnn_edge_kernel<<<EE / 128, blk, 0, stream>>>(
        fin, pin, edge_attr, rowI, colI,
        msg_W1 + (long long)l * 131 * 64, msg_b1 + l * 64,
        msg_W2 + (long long)l * 64 * 64,  msg_b2 + l * 64,
        pos_W1 + (long long)l * 64 * 64,  pos_b1 + l * 64,
        pos_W2 + (long long)l * 64,       pos_b2 + l,
        msgAgg, posAgg);

    _Float16* fout = (fin == featA) ? featB : featA;
    float* po = (l == 4) ? (float*)d_out : ((l & 1) ? posB : posA);

    egnn_node_kernel<<<(NN + 127) / 128, blk, 0, stream>>>(
        fin, msgAgg, invC, pin, posAgg,
        nf_W1 + (long long)l * 128 * 64, nf_b1 + l * 64,
        nf_W2 + (long long)l * 64 * 64,  nf_b2 + l * 64,
        fout, po, (l == 0) ? 0 : 1);   // layer 0 (com_node_feat_net) keeps pos

    fin = fout;
    pin = po;
  }
}
